// BACA_84301618086454
// MI455X (gfx1250) — matrix operand order repaired and checked by reading, not compiled, not run on hardware
//
#include <hip/hip_runtime.h>
#include <hip/hip_bf16.h>
#include <math.h>

// ---------------- static config (matches reference) ----------------
#define DIMC   256
#define CR     64
#define GC     32
#define NH     4
#define HC     16
#define GROUP  2
#define KS     3
#define NSAMP  9          // KS*KS
#define NUMCLS 80
#define BATCH  8
#define HH     64
#define WW     64
#define HWSZ   4096       // HH*WW
#define NPIX   32768      // BATCH*HWSZ
#define QKVCH  192        // 3*CR
#define MULF   5.0f
#define SCALEF 0.25f      // HC^-0.5

typedef __attribute__((ext_vector_type(16))) _Float16 v16h;
typedef __attribute__((ext_vector_type(8)))  float    v8f;

// ------------------------------------------------------------------
// Register-blocked WMMA accumulate: MB x NB tiles of 16x16.
//   A row-major [M][lda] f16, rows = out-channels
//   Bt row-major [N][ldb] f16, rows = pixels
// Fragment packing per documented CDNA5 wave32 16-bit layouts:
//   A 16x32: lanes0-15 K in {h*8+2q | q<4} U {16+h*8+2q}, h = lane>>4
//   B 32x16: same K order as A
// ------------------------------------------------------------------
template<int MB, int NB>
__device__ __forceinline__ void wmma_block_acc(const _Float16* __restrict__ A, int lda,
                                               const _Float16* __restrict__ Bt, int ldb,
                                               int mrow0, int nrow0, int K,
                                               v8f acc[MB][NB]) {
    const int lane = threadIdx.x & 31;
    const int half = lane >> 4;
    const int l15  = lane & 15;
    const _Float16* arow[MB];
    const _Float16* brow[NB];
#pragma unroll
    for (int i = 0; i < MB; ++i) arow[i] = A  + (size_t)(mrow0 + i * 16 + l15) * lda;
#pragma unroll
    for (int j = 0; j < NB; ++j) brow[j] = Bt + (size_t)(nrow0 + j * 16 + l15) * ldb;

    for (int k0 = 0; k0 < K; k0 += 32) {
        union { v16h v; unsigned u[8]; } a[MB], b[NB];
#pragma unroll
        for (int i = 0; i < MB; ++i) {
#pragma unroll
            for (int q = 0; q < 4; ++q) {
                a[i].u[q]     = *(const unsigned*)(arow[i] + k0 + half * 8 + 2 * q);
                a[i].u[4 + q] = *(const unsigned*)(arow[i] + k0 + 16 + half * 8 + 2 * q);
            }
        }
#pragma unroll
        for (int j = 0; j < NB; ++j)
#pragma unroll
            for (int q = 0; q < 8; ++q)
                b[j].u[q] = *(const unsigned*)(brow[j] + k0 + (q < 4 ? half * 8 + 2 * q : 16 + half * 8 + 2 * (q - 4)));
#pragma unroll
        for (int i = 0; i < MB; ++i)
#pragma unroll
            for (int j = 0; j < NB; ++j)
                acc[i][j] = __builtin_amdgcn_wmma_f32_16x16x32_f16(false, a[i].v, false, b[j].v,
                                                                   (short)0, acc[i][j], false, false);
    }
}

// ---------------- kernel 0: f32 -> f16 convert --------------------
__global__ void cvt_f16_kernel(const float* __restrict__ src, _Float16* __restrict__ dst, int n) {
    int i = blockIdx.x * blockDim.x + threadIdx.x;
    if (i < n) dst[i] = (_Float16)src[i];
}

// ---------------- kernel 1: LayerNorm over 256 ch, emit f16 -------
// one wave per pixel, lane covers channels lane+32*i
__global__ void ln1_kernel(const float* __restrict__ x, const float* __restrict__ g,
                           const float* __restrict__ bt, _Float16* __restrict__ xln) {
    int wid  = (blockIdx.x * blockDim.x + threadIdx.x) >> 5;   // pixel p
    int lane = threadIdx.x & 31;
    int b = wid >> 12, hw = wid & 4095;
    const float* base = x + (((size_t)b * DIMC) << 12) + hw;
    float v[8];
    float s = 0.f;
#pragma unroll
    for (int i = 0; i < 8; ++i) {
        v[i] = base[((size_t)(lane + 32 * i)) << 12];
        s += v[i];
    }
#pragma unroll
    for (int m = 1; m < 32; m <<= 1) s += __shfl_xor(s, m, 32);
    float mean = s * (1.f / 256.f);
    float vs = 0.f;
#pragma unroll
    for (int i = 0; i < 8; ++i) { float d = v[i] - mean; vs += d * d; }
#pragma unroll
    for (int m = 1; m < 32; m <<= 1) vs += __shfl_xor(vs, m, 32);
    float rs = rsqrtf(vs * (1.f / 256.f) + 1e-5f);
    _Float16* orow = xln + (size_t)wid * DIMC;
#pragma unroll
    for (int i = 0; i < 8; ++i) {
        int c = lane + 32 * i;
        orow[c] = (_Float16)((v[i] - mean) * rs * g[c] + bt[c]);
    }
}

// ---------------- kernel 2: qkv GEMM (WMMA, 2x4 blocked) ----------
// out[o][p] = sum_c wqkv[o][c] * xln[p][c], stored as [B,192,H,W] f32
__global__ void gemm_qkv_kernel(const _Float16* __restrict__ A, const _Float16* __restrict__ Bt,
                                float* __restrict__ out) {
    int wid = (blockIdx.x * blockDim.x + threadIdx.x) >> 5;
    const int MG = (QKVCH / 16) / 2;           // 6 mtile-groups of 2
    int mg = wid % MG, ng = wid / MG;
    v8f acc[2][4] = {};
    wmma_block_acc<2, 4>(A, DIMC, Bt, DIMC, mg * 32, ng * 64, DIMC, acc);
    int lane = threadIdx.x & 31, half = lane >> 4, l15 = lane & 15;
#pragma unroll
    for (int j = 0; j < 4; ++j) {
        int p = ng * 64 + j * 16 + l15;
        int b = p >> 12, hw = p & 4095;
#pragma unroll
        for (int i = 0; i < 2; ++i)
#pragma unroll
            for (int r = 0; r < 8; ++r) {
                int o = mg * 32 + i * 16 + r + 8 * half;
                out[(((size_t)(b * QKVCH + o)) << 12) + hw] = acc[i][j][r];
            }
    }
}

// ---------------- kernel 3: depthwise3x3 + LN(32) + GELU ----------
// one wave per (bg, pixel), lane = channel 0..31
__global__ void dwln_kernel(const float* __restrict__ qkv, const float* __restrict__ wdw,
                            const float* __restrict__ g2, const float* __restrict__ b2,
                            float* __restrict__ t) {
    int wid  = (blockIdx.x * blockDim.x + threadIdx.x) >> 5;   // bg*4096 + hw
    int lane = threadIdx.x & 31;
    int bg = wid >> 12, hw = wid & 4095;
    int b = bg >> 1, gg = bg & 1;
    int h = hw >> 6, w = hw & 63;
    const float* plane = qkv + (((size_t)(b * QKVCH + gg * GC + lane)) << 12);
    const float* wk = wdw + lane * 9;
    float acc = 0.f;
#pragma unroll
    for (int dy = -1; dy <= 1; ++dy)
#pragma unroll
        for (int dx = -1; dx <= 1; ++dx) {
            int yy = h + dy, xx = w + dx;
            if (yy >= 0 && yy < HH && xx >= 0 && xx < WW)
                acc += plane[yy * WW + xx] * wk[(dy + 1) * 3 + (dx + 1)];
        }
    float s = acc;
#pragma unroll
    for (int m = 1; m < 32; m <<= 1) s += __shfl_xor(s, m, 32);
    float mean = s * (1.f / 32.f);
    float d = acc - mean;
    float vs = d * d;
#pragma unroll
    for (int m = 1; m < 32; m <<= 1) vs += __shfl_xor(vs, m, 32);
    float rs = rsqrtf(vs * (1.f / 32.f) + 1e-5f);
    float y = d * rs * g2[lane] + b2[lane];
    y = 0.5f * y * (1.f + erff(y * 0.70710678118654752f));     // exact GELU
    t[(((size_t)(bg * GC + lane)) << 12) + hw] = y;
}

// ---------------- kernel 4: 3x3 conv 32->18 + tanh*5 + base -------
__global__ void offconv_kernel(const float* __restrict__ t, const float* __restrict__ woff,
                               const float* __restrict__ boff, const float* __restrict__ base_off,
                               float* __restrict__ off) {
    int idx = blockIdx.x * blockDim.x + threadIdx.x;           // (bg*18+oc)*4096 + hw
    if (idx >= 16 * 18 * HWSZ) return;
    int hw = idx & 4095;
    int tmp = idx >> 12;
    int oc = tmp % 18;
    int bg = tmp / 18;
    int b = bg >> 1;
    int h = hw >> 6, w = hw & 63;
    float acc = boff[oc];
    const float* wb = woff + oc * GC * 9;
    for (int c = 0; c < GC; ++c) {
        const float* plane = t + (((size_t)(bg * GC + c)) << 12);
        const float* wc = wb + c * 9;
#pragma unroll
        for (int dy = -1; dy <= 1; ++dy)
#pragma unroll
            for (int dx = -1; dx <= 1; ++dx) {
                int yy = h + dy, xx = w + dx;
                if (yy >= 0 && yy < HH && xx >= 0 && xx < WW)
                    acc += plane[yy * WW + xx] * wc[(dy + 1) * 3 + (dx + 1)];
            }
    }
    float base = base_off[(((size_t)(b * 18 + oc)) << 12) + hw];
    off[idx] = tanhf(acc) * MULF + base;
}

// ---------------- kernel 5: deformable gather + attention ---------
// wave = (b, g, pixel); lane = head_local*16 + hc; feature channel = lane
__global__ void attn_kernel(const float* __restrict__ qkv, const float* __restrict__ off,
                            const float* __restrict__ rpb, _Float16* __restrict__ att) {
    int wid  = (blockIdx.x * blockDim.x + threadIdx.x) >> 5;   // bg*4096 + hw
    int lane = threadIdx.x & 31;
    int bg = wid >> 12, hw = wid & 4095;
    int b = bg >> 1, gg = bg & 1;
    int head = gg * 2 + (lane >> 4);
    int hc   = lane & 15;

    const float* kp = qkv + (((size_t)(b * QKVCH + CR + gg * GC + lane)) << 12);
    const float* vp = kp + (((size_t)CR) << 12);               // v plane = k plane + 64 ch
    const float* qp = qkv + (((size_t)(b * QKVCH + gg * GC + lane)) << 12);
    const float* ob = off + (((size_t)bg * 18) << 12) + hw;

    float qv = qp[hw] * SCALEF;

    float lg[NSAMP], vr[NSAMP];
#pragma unroll
    for (int n = 0; n < NSAMP; ++n) {
        float oy = ob[((size_t)(2 * n)) << 12];
        float ox = ob[((size_t)(2 * n + 1)) << 12];
        float x0f = floorf(ox), y0f = floorf(oy);
        float fx = ox - x0f, fy = oy - y0f;
        int ix0 = (int)x0f, iy0 = (int)y0f;
        float kk = 0.f, vv = 0.f;
#pragma unroll
        for (int ty = 0; ty < 2; ++ty)
#pragma unroll
            for (int tx = 0; tx < 2; ++tx) {
                int xx = ix0 + tx, yy = iy0 + ty;
                float wt = (tx ? fx : (1.f - fx)) * (ty ? fy : (1.f - fy));
                if (xx >= 0 && xx < WW && yy >= 0 && yy < HH) {
                    int a = yy * WW + xx;
                    kk += kp[a] * wt;
                    vv += vp[a] * wt;
                }
            }
        kk += rpb[(head * NSAMP + n) * HC + hc];
        float l = qv * kk;
        l += __shfl_xor(l, 1, 32);
        l += __shfl_xor(l, 2, 32);
        l += __shfl_xor(l, 4, 32);
        l += __shfl_xor(l, 8, 32);                             // reduce 16-lane half (hc)
        lg[n] = l;
        vr[n] = vv;
    }
    float mx = lg[0];
#pragma unroll
    for (int n = 1; n < NSAMP; ++n) mx = fmaxf(mx, lg[n]);
    float s = 0.f, o = 0.f;
#pragma unroll
    for (int n = 0; n < NSAMP; ++n) {
        float e = __expf(lg[n] - mx);
        s += e;
        o += e * vr[n];
    }
    o /= s;
    int p = (b << 12) | hw;
    att[(size_t)p * CR + gg * GC + lane] = (_Float16)o;        // [p][cr]
}

// ---------------- kernel 6: proj GEMM + bias + residual (WMMA 2x4) 
// y[p][c] = att[p][:]@wproj[c][:] + bproj[c] + x[b][c][hw], stored f16
__global__ void gemm_proj_kernel(const _Float16* __restrict__ A, const _Float16* __restrict__ Bt,
                                 const float* __restrict__ bproj, const float* __restrict__ x,
                                 _Float16* __restrict__ y) {
    int wid = (blockIdx.x * blockDim.x + threadIdx.x) >> 5;
    const int MG = (DIMC / 16) / 2;            // 8 mtile-groups of 2
    int mg = wid % MG, ng = wid / MG;
    v8f acc[2][4] = {};
    wmma_block_acc<2, 4>(A, CR, Bt, CR, mg * 32, ng * 64, CR, acc);
    int lane = threadIdx.x & 31, half = lane >> 4, l15 = lane & 15;
#pragma unroll
    for (int j = 0; j < 4; ++j) {
        int p = ng * 64 + j * 16 + l15;
        int b = p >> 12, hw = p & 4095;
#pragma unroll
        for (int i = 0; i < 2; ++i)
#pragma unroll
            for (int r = 0; r < 8; ++r) {
                int c = mg * 32 + i * 16 + r + 8 * half;
                float res = x[(((size_t)(b * DIMC + c)) << 12) + hw];
                y[(size_t)p * DIMC + c] = (_Float16)(acc[i][j][r] + bproj[c] + res);
            }
    }
}

// ---------------- kernel 7: classifier GEMM + bias (WMMA 1x4) -----
// d_out[b][cls][hw] = y[p][:]@wcls[cls][:] + bcls[cls]
__global__ void gemm_cls_kernel(const _Float16* __restrict__ A, const _Float16* __restrict__ Bt,
                                const float* __restrict__ bcls, float* __restrict__ out) {
    int wid = (blockIdx.x * blockDim.x + threadIdx.x) >> 5;
    const int MG = NUMCLS / 16;                // 5 mtiles (MB=1)
    int mg = wid % MG, ng = wid / MG;
    v8f acc[1][4] = {};
    wmma_block_acc<1, 4>(A, DIMC, Bt, DIMC, mg * 16, ng * 64, DIMC, acc);
    int lane = threadIdx.x & 31, half = lane >> 4, l15 = lane & 15;
#pragma unroll
    for (int j = 0; j < 4; ++j) {
        int p = ng * 64 + j * 16 + l15;
        int b = p >> 12, hw = p & 4095;
#pragma unroll
        for (int r = 0; r < 8; ++r) {
            int o = mg * 16 + r + 8 * half;
            out[(((size_t)(b * NUMCLS + o)) << 12) + hw] = acc[0][j][r] + bcls[o];
        }
    }
}

// ------------------------------------------------------------------
extern "C" void kernel_launch(void* const* d_in, const int* in_sizes, int n_in,
                              void* d_out, int out_size, void* d_ws, size_t ws_size,
                              hipStream_t stream) {
    (void)in_sizes; (void)n_in; (void)out_size; (void)ws_size;
    const float* x       = (const float*)d_in[0];
    const float* offset  = (const float*)d_in[1];
    const float* ln1_g   = (const float*)d_in[2];
    const float* ln1_b   = (const float*)d_in[3];
    const float* w_qkv   = (const float*)d_in[4];
    const float* w_dw    = (const float*)d_in[5];
    const float* ln2_g   = (const float*)d_in[6];
    const float* ln2_b   = (const float*)d_in[7];
    const float* w_off   = (const float*)d_in[8];
    const float* b_off   = (const float*)d_in[9];
    const float* rpb     = (const float*)d_in[10];
    const float* w_proj  = (const float*)d_in[11];
    const float* b_proj  = (const float*)d_in[12];
    const float* w_cls   = (const float*)d_in[13];
    const float* b_cls   = (const float*)d_in[14];
    float* out = (float*)d_out;

    char* ws = (char*)d_ws;
    _Float16* xln_h   = (_Float16*)(ws + 0);                    // 32768*256 f16 = 16 MiB
    float*    qkv_f   = (float*)   (ws + 16777216);             // 8*192*4096 f32 = 24 MiB
    float*    t_f     = (float*)   (ws + 41943040);             // 16*32*4096 f32 = 8 MiB
    float*    off_f   = (float*)   (ws + 50331648);             // 16*18*4096 f32 = 4.5 MiB
    _Float16* att_h   = (_Float16*)(ws + 55050240);             // 32768*64 f16 = 4 MiB
    _Float16* y_h     = (_Float16*)(ws + 59244544);             // 32768*256 f16 = 16 MiB
    _Float16* wqkv_h  = (_Float16*)(ws + 76021760);             // 192*256
    _Float16* wproj_h = (_Float16*)(ws + 76120064);             // 256*64
    _Float16* wcls_h  = (_Float16*)(ws + 76152832);             // 80*256

    // weight conversions
    cvt_f16_kernel<<<(QKVCH * DIMC + 255) / 256, 256, 0, stream>>>(w_qkv, wqkv_h, QKVCH * DIMC);
    cvt_f16_kernel<<<(DIMC * CR + 255) / 256, 256, 0, stream>>>(w_proj, wproj_h, DIMC * CR);
    cvt_f16_kernel<<<(NUMCLS * DIMC + 255) / 256, 256, 0, stream>>>(w_cls, wcls_h, NUMCLS * DIMC);

    // 1. LN over channels -> f16 pixel-major activations
    ln1_kernel<<<NPIX / 8, 256, 0, stream>>>(x, ln1_g, ln1_b, xln_h);

    // 2. qkv GEMM (WMMA 2x4): 6 mgroups * 512 ngroups = 3072 waves
    gemm_qkv_kernel<<<(6 * (NPIX / 64)) / 8, 256, 0, stream>>>(wqkv_h, xln_h, qkv_f);

    // 3. depthwise 3x3 + LN(32) + GELU
    dwln_kernel<<<(16 * HWSZ) / 8, 256, 0, stream>>>(qkv_f, w_dw, ln2_g, ln2_b, t_f);

    // 4. offset conv 32->18, tanh*5 + base
    offconv_kernel<<<(16 * 18 * HWSZ + 255) / 256, 256, 0, stream>>>(t_f, w_off, b_off, offset, off_f);

    // 5. deformable gather + attention
    attn_kernel<<<(16 * HWSZ) / 8, 256, 0, stream>>>(qkv_f, off_f, rpb, att_h);

    // 6. proj GEMM + bias + residual (WMMA 2x4): 8 * 512 = 4096 waves
    gemm_proj_kernel<<<(8 * (NPIX / 64)) / 8, 256, 0, stream>>>(wproj_h, att_h, b_proj, x, y_h);

    // 7. classifier GEMM + bias (WMMA 1x4): 5 * 512 = 2560 waves
    gemm_cls_kernel<<<(5 * (NPIX / 64)) / 8, 256, 0, stream>>>(wcls_h, y_h, b_cls, out);
}
